// LoRA_MOE_QK_old_28381143892013
// MI455X (gfx1250) — matrix operand order repaired and checked by reading, not compiled, not run on hardware
//
#include <hip/hip_runtime.h>
#include <hip/hip_bf16.h>

// Problem constants
#define Bq  8
#define Sq  2048
#define Dq  4096
#define Oq  4096
#define Eq  8
#define Rq  16
#define IMG_START 34
#define IMG_LEN   576
#define SCALING   2.0f
#define Mq  (Bq*Sq)          // 16384
#define MIMG (Bq*IMG_LEN)    // 4608
#define DU  (Dq/2)           // row stride of bf16 matrices in u32 units

typedef unsigned int u32;
typedef __attribute__((ext_vector_type(16))) __bf16 v16bf;
typedef __attribute__((ext_vector_type(8)))  float  v8f;
typedef __attribute__((ext_vector_type(4)))  int    v4i;

__device__ __forceinline__ u32 f2bf(float f) {
    u32 u = __float_as_uint(f);
    u32 r = u + 0x7FFFu + ((u >> 16) & 1u);   // round-to-nearest-even
    return r >> 16;
}
__device__ __forceinline__ u32 pack2(float lo, float hi) {
    return f2bf(lo) | (f2bf(hi) << 16);
}

union Frag { uint4 q[2]; v16bf v; };

// CDNA5 async global->LDS copy (ASYNCcnt path), 16 bytes per lane.
// clang-22 declares the builtin with v4i* parameters (per compile diagnostic).
__device__ __forceinline__ void async_copy16(const u32* g, u32* l) {
    __builtin_amdgcn_global_load_async_to_lds_b128((v4i*)g, (v4i*)l, 0, 0);
}

// ---------------------------------------------------------------------------
// f32 -> bf16 pre-conversion (one uint4 = 8 elements per thread)
// ---------------------------------------------------------------------------
__global__ __launch_bounds__(256) void cvt_f32_bf16_kernel(
    const float* __restrict__ src, u32* __restrict__ dst)
{
    const long i = (long)blockIdx.x * 256 + threadIdx.x;
    const float4* s = (const float4*)src;
    const float4 a = s[2 * i], b = s[2 * i + 1];
    uint4 q = { pack2(a.x, a.y), pack2(a.z, a.w),
                pack2(b.x, b.y), pack2(b.z, b.w) };
    ((uint4*)dst)[i] = q;
}

// ---------------------------------------------------------------------------
// Main GEMM (async path): final[m,o] = xb[m,:] . wb[o,:] + bias[o]
// bf16 operands pre-staged in workspace; double-buffered LDS tiles filled by
// global_load_async_to_lds_b128, consumed by v_wmma_f32_16x16x32_bf16.
// block tile 128x128, BK=32, 8 waves, each wave 64x32 (4x2 wmma tiles)
// ---------------------------------------------------------------------------
__global__ __launch_bounds__(256) void gemm_bias_async_kernel(
    const u32* __restrict__ xb, const u32* __restrict__ wb,
    const float* __restrict__ bias, float* __restrict__ out)
{
    // padded stride 20 dwords -> conflict-free fragment reads, 16B aligned
    __shared__ u32 At[2][128 * 20];
    __shared__ u32 Bt[2][128 * 20];

    const int t    = threadIdx.x;
    const int lane = t & 31;
    const int wave = t >> 5;
    const int lr   = lane & 15;
    const int hi   = lane >> 4;
    const int wm   = (wave & 1) * 64;
    const int wn   = (wave >> 1) * 32;
    const long m0  = (long)blockIdx.y * 128;
    const long n0  = (long)blockIdx.x * 128;

    // staging: each thread copies 32B of A and 32B of B per K-step
    const int lrow = t >> 1;          // 0..127
    const int khu  = (t & 1) * 8;     // u32 offset inside 16-u32 (BK=32) row
    const u32* ga = xb + (m0 + lrow) * (long)DU + khu;
    const u32* gb = wb + (n0 + lrow) * (long)DU + khu;
    const int lofs = lrow * 20 + khu;

    v8f acc[4][2];
    #pragma unroll
    for (int i = 0; i < 4; ++i)
        #pragma unroll
        for (int j = 0; j < 2; ++j)
            #pragma unroll
            for (int v = 0; v < 8; ++v) acc[i][j][v] = 0.0f;

    auto stage = [&](int buf, int k0) {
        const u32* a = ga + (k0 >> 1);
        const u32* b = gb + (k0 >> 1);
        async_copy16(a,     &At[buf][lofs]);
        async_copy16(a + 4, &At[buf][lofs + 4]);
        async_copy16(b,     &Bt[buf][lofs]);
        async_copy16(b + 4, &Bt[buf][lofs + 4]);
    };

    stage(0, 0);
    __builtin_amdgcn_s_wait_asynccnt(0);
    __syncthreads();

    int buf = 0;
    for (int k0 = 0; k0 < Dq; k0 += 32) {
        if (k0 + 32 < Dq) stage(buf ^ 1, k0 + 32);   // overlap next tile DMA

        // fragments per ISA 7.12.2 bf16 layouts
        Frag af[4];
        #pragma unroll
        for (int mi = 0; mi < 4; ++mi) {
            const u32* base = &At[buf][(wm + mi * 16 + lr) * 20];
            af[mi].q[0] = *(const uint4*)(base + hi * 4);
            af[mi].q[1] = *(const uint4*)(base + hi * 4 + 8);
        }
        Frag bf[2];
        #pragma unroll
        for (int ni = 0; ni < 2; ++ni) {
            const u32* base = &Bt[buf][(wn + ni * 16 + lr) * 20];
            bf[ni].q[0] = *(const uint4*)(base + hi * 4);
            bf[ni].q[1] = *(const uint4*)(base + hi * 4 + 8);
        }

        #pragma unroll
        for (int mi = 0; mi < 4; ++mi)
            #pragma unroll
            for (int ni = 0; ni < 2; ++ni)
                acc[mi][ni] = __builtin_amdgcn_wmma_f32_16x16x32_bf16(
                    false, af[mi].v, false, bf[ni].v,
                    (short)0, acc[mi][ni], false, false);

        __builtin_amdgcn_s_wait_asynccnt(0);
        __syncthreads();
        buf ^= 1;
    }

    #pragma unroll
    for (int ni = 0; ni < 2; ++ni) {
        const long col = n0 + wn + ni * 16 + lr;
        const float bv = bias[col];
        #pragma unroll
        for (int mi = 0; mi < 4; ++mi) {
            #pragma unroll
            for (int v = 0; v < 8; ++v) {
                const long row = m0 + wm + mi * 16 + hi * 8 + v;
                out[row * (long)Oq + col] = acc[mi][ni][v] + bv;
            }
        }
    }
}

// ---------------------------------------------------------------------------
// Fallback GEMM (no workspace): fused f32->bf16 conversion while staging.
// ---------------------------------------------------------------------------
__global__ __launch_bounds__(256) void gemm_bias_kernel(
    const float* __restrict__ x, const float* __restrict__ W,
    const float* __restrict__ bias, float* __restrict__ out)
{
    __shared__ u32 At[128 * 20];
    __shared__ u32 Bt[128 * 20];

    const int t    = threadIdx.x;
    const int lane = t & 31;
    const int wave = t >> 5;
    const int lr   = lane & 15;
    const int hi   = lane >> 4;
    const int wm   = (wave & 1) * 64;
    const int wn   = (wave >> 1) * 32;
    const long m0  = (long)blockIdx.y * 128;
    const long n0  = (long)blockIdx.x * 128;

    const int lrow = t >> 1;
    const int kh   = (t & 1) * 16;
    const float* ag = x + (m0 + lrow) * (long)Dq + kh;
    const float* bg = W + (n0 + lrow) * (long)Dq + kh;
    u32* aw = &At[lrow * 20 + (kh >> 1)];
    u32* bw = &Bt[lrow * 20 + (kh >> 1)];

    v8f acc[4][2];
    #pragma unroll
    for (int i = 0; i < 4; ++i)
        #pragma unroll
        for (int j = 0; j < 2; ++j)
            #pragma unroll
            for (int v = 0; v < 8; ++v) acc[i][j][v] = 0.0f;

    for (int k0 = 0; k0 < Dq; k0 += 32) {
        const float4* agv = (const float4*)(ag + k0);
        const float4* bgv = (const float4*)(bg + k0);
        float4 a4[4], b4[4];
        #pragma unroll
        for (int i = 0; i < 4; ++i) { a4[i] = agv[i]; b4[i] = bgv[i]; }

        if (k0 + 32 < Dq) {
            __builtin_prefetch(ag + k0 + 32, 0, 3);
            __builtin_prefetch(bg + k0 + 32, 0, 3);
        }

        uint4 qa0 = { pack2(a4[0].x, a4[0].y), pack2(a4[0].z, a4[0].w),
                      pack2(a4[1].x, a4[1].y), pack2(a4[1].z, a4[1].w) };
        uint4 qa1 = { pack2(a4[2].x, a4[2].y), pack2(a4[2].z, a4[2].w),
                      pack2(a4[3].x, a4[3].y), pack2(a4[3].z, a4[3].w) };
        uint4 qb0 = { pack2(b4[0].x, b4[0].y), pack2(b4[0].z, b4[0].w),
                      pack2(b4[1].x, b4[1].y), pack2(b4[1].z, b4[1].w) };
        uint4 qb1 = { pack2(b4[2].x, b4[2].y), pack2(b4[2].z, b4[2].w),
                      pack2(b4[3].x, b4[3].y), pack2(b4[3].z, b4[3].w) };
        *(uint4*)(aw + 0) = qa0;  *(uint4*)(aw + 4) = qa1;
        *(uint4*)(bw + 0) = qb0;  *(uint4*)(bw + 4) = qb1;
        __syncthreads();

        Frag af[4];
        #pragma unroll
        for (int mi = 0; mi < 4; ++mi) {
            const u32* base = &At[(wm + mi * 16 + lr) * 20];
            af[mi].q[0] = *(const uint4*)(base + hi * 4);
            af[mi].q[1] = *(const uint4*)(base + hi * 4 + 8);
        }
        Frag bf[2];
        #pragma unroll
        for (int ni = 0; ni < 2; ++ni) {
            const u32* base = &Bt[(wn + ni * 16 + lr) * 20];
            bf[ni].q[0] = *(const uint4*)(base + hi * 4);
            bf[ni].q[1] = *(const uint4*)(base + hi * 4 + 8);
        }

        #pragma unroll
        for (int mi = 0; mi < 4; ++mi)
            #pragma unroll
            for (int ni = 0; ni < 2; ++ni)
                acc[mi][ni] = __builtin_amdgcn_wmma_f32_16x16x32_bf16(
                    false, af[mi].v, false, bf[ni].v,
                    (short)0, acc[mi][ni], false, false);
        __syncthreads();
    }

    #pragma unroll
    for (int ni = 0; ni < 2; ++ni) {
        const long col = n0 + wn + ni * 16 + lr;
        const float bv = bias[col];
        #pragma unroll
        for (int mi = 0; mi < 4; ++mi) {
            #pragma unroll
            for (int v = 0; v < 8; ++v) {
                const long row = m0 + wm + mi * 16 + hi * 8 + v;
                out[row * (long)Oq + col] = acc[mi][ni][v] + bv;
            }
        }
    }
}

// ---------------------------------------------------------------------------
// Routing: logits -> softmax -> argmax.  One wave per image row.
// ---------------------------------------------------------------------------
__global__ __launch_bounds__(256) void routing_kernel(
    const float* __restrict__ x, const float* __restrict__ Wr,
    const float* __restrict__ br, float* __restrict__ routing_out,
    float* __restrict__ choice_out, int* __restrict__ idx_out)
{
    const int row  = blockIdx.x * 8 + (threadIdx.x >> 5);
    const int lane = threadIdx.x & 31;
    const int b = row / IMG_LEN, s = row % IMG_LEN;
    const float* xr = x + ((long)(b * Sq + IMG_START + s)) * Dq;

    float acc[Eq];
    #pragma unroll
    for (int e = 0; e < Eq; ++e) acc[e] = 0.0f;

    for (int i = 0; i < Dq / 32; ++i) {
        const int d = lane + 32 * i;
        const float xv = xr[d];
        #pragma unroll
        for (int e = 0; e < Eq; ++e) acc[e] += xv * Wr[e * Dq + d];
    }
    #pragma unroll
    for (int off = 16; off > 0; off >>= 1)
        #pragma unroll
        for (int e = 0; e < Eq; ++e) acc[e] += __shfl_xor(acc[e], off, 32);

    if (lane == 0) {
        float lg[Eq], p[Eq];
        float mx = -3.0e38f; int ai = 0;
        #pragma unroll
        for (int e = 0; e < Eq; ++e) {
            lg[e] = acc[e] + br[e];
            if (lg[e] > mx) { mx = lg[e]; ai = e; }   // first max, like argmax
        }
        float sum = 0.0f;
        #pragma unroll
        for (int e = 0; e < Eq; ++e) { p[e] = __expf(lg[e] - mx); sum += p[e]; }
        const float inv = 1.0f / sum;
        #pragma unroll
        for (int e = 0; e < Eq; ++e) {
            const float r  = p[e] * inv;
            const float oh = (e == ai) ? 1.0f : 0.0f;
            routing_out[(long)row * Eq + e] = r;
            choice_out [(long)row * Eq + e] = (oh - r) + r;  // forward == onehot
        }
        idx_out[row] = ai;
    }
}

// ---------------------------------------------------------------------------
// LoRA stage 1: u[row, r] = x_img[row, :] . A[e*, r, :]   (one block per row)
// ---------------------------------------------------------------------------
__global__ __launch_bounds__(256) void lora_u_kernel(
    const float* __restrict__ x, const float* __restrict__ A,
    const int* __restrict__ idx, float* __restrict__ u)
{
    const int row = blockIdx.x;
    const int t   = threadIdx.x;
    const int r   = t >> 4, l16 = t & 15;
    const int b = row / IMG_LEN, s = row % IMG_LEN;
    const float* xr = x + ((long)(b * Sq + IMG_START + s)) * Dq;
    const int e = idx[row];
    const float* Ar = A + ((long)e * Rq + r) * Dq;

    float sum = 0.0f;
    for (int i = 0; i < Dq / 16; ++i) {
        const int d = l16 + 16 * i;
        sum += xr[d] * Ar[d];
    }
    #pragma unroll
    for (int off = 8; off > 0; off >>= 1) sum += __shfl_down(sum, off, 16);
    if (l16 == 0) u[(long)row * Rq + r] = sum;
}

// ---------------------------------------------------------------------------
// LoRA stage 2: final[img row, o] += SCALING * u[row,:] . Bm[e*, o, :]
// ---------------------------------------------------------------------------
__global__ __launch_bounds__(256) void lora_add_kernel(
    const float* __restrict__ u, const float* __restrict__ Bm,
    const int* __restrict__ idx, float* __restrict__ out)
{
    const int o   = blockIdx.x * 256 + threadIdx.x;
    const int row = blockIdx.y;
    const int e   = idx[row];
    const float* ur = u + (long)row * Rq;
    const float* Br = Bm + ((long)e * Oq + o) * Rq;

    float s = 0.0f;
    #pragma unroll
    for (int r = 0; r < Rq; ++r) s += ur[r] * Br[r];

    const int b = row / IMG_LEN, sp = row % IMG_LEN;
    const long fi = ((long)(b * Sq + IMG_START + sp)) * Oq + o;
    out[fi] += s * SCALING;
}

// ---------------------------------------------------------------------------
extern "C" void kernel_launch(void* const* d_in, const int* in_sizes, int n_in,
                              void* d_out, int out_size, void* d_ws, size_t ws_size,
                              hipStream_t stream) {
    const float* x    = (const float*)d_in[0];
    const float* W    = (const float*)d_in[1];
    const float* bias = (const float*)d_in[2];
    const float* A    = (const float*)d_in[3];
    const float* Bm   = (const float*)d_in[4];
    const float* Wr   = (const float*)d_in[5];
    const float* br   = (const float*)d_in[6];

    float* final_out   = (float*)d_out;
    float* routing_out = final_out + (size_t)Mq * Oq;
    float* choice_out  = routing_out + (size_t)MIMG * Eq;

    int*   idx = (int*)d_ws;
    float* u   = (float*)((char*)d_ws + 32768);

    const size_t XB_OFF   = (size_t)1 << 20;            // small-scratch region
    const size_t XB_BYTES = (size_t)Mq * Dq * 2;        // 128 MiB bf16 x
    const size_t WB_BYTES = (size_t)Oq * Dq * 2;        //  32 MiB bf16 W
    const size_t NEED     = XB_OFF + XB_BYTES + WB_BYTES;

    if (ws_size >= NEED) {
        u32* xbw = (u32*)((char*)d_ws + XB_OFF);
        u32* wbw = (u32*)((char*)d_ws + XB_OFF + XB_BYTES);
        // one-time f32->bf16 conversion (~480 MiB total traffic)
        cvt_f32_bf16_kernel<<<(Mq * (Dq / 8)) / 256, 256, 0, stream>>>(x, xbw);
        cvt_f32_bf16_kernel<<<(Oq * (Dq / 8)) / 256, 256, 0, stream>>>(W, wbw);
        gemm_bias_async_kernel<<<dim3(Oq / 128, Mq / 128), 256, 0, stream>>>(
            xbw, wbw, bias, final_out);
    } else {
        gemm_bias_kernel<<<dim3(Oq / 128, Mq / 128), 256, 0, stream>>>(
            x, W, bias, final_out);
    }

    routing_kernel<<<MIMG / 8, 256, 0, stream>>>(x, Wr, br, routing_out, choice_out, idx);
    lora_u_kernel<<<MIMG, 256, 0, stream>>>(x, A, idx, u);
    lora_add_kernel<<<dim3(Oq / 256, MIMG), 256, 0, stream>>>(u, Bm, idx, final_out);
}
